// FrameTransformer_68917045231700
// MI455X (gfx1250) — compile-verified, matrix operand order checked by reading, not run on hardware
//
#include <hip/hip_runtime.h>
#include <hip/hip_bf16.h>
#include <math.h>
#include <stdint.h>

// ---------------------------------------------------------------------------
// CDNA5 / gfx1250 implementation of the FrameTransformer U-Net.
// Convs, the 32x32 mc_pw and the channel mix run on v_wmma_f32_16x16x32_f16 (f16 operands, f32
// accumulate).  Conv weights are pre-packed once per conv into f16 in exact
// WMMA B-fragment order so the inner loop loads a whole fragment with one
// 32-byte vector load.  Instance-norm is folded into a per-instance affine
// applied inside the consumer conv's A-fragment load; leaky-ReLU is folded
// into the next conv's input read.  Attention K/V/Q tiles are staged into LDS
// with the Tensor Data Mover (tensor_load_to_lds + s_wait_tensorcnt).
// wave32 throughout.
// ---------------------------------------------------------------------------

typedef __attribute__((ext_vector_type(16))) _Float16 v16h;
typedef __attribute__((ext_vector_type(8)))  float    v8f;
typedef __attribute__((ext_vector_type(4)))  unsigned int v4u;
typedef __attribute__((ext_vector_type(8)))  int      v8i;
typedef __attribute__((ext_vector_type(4)))  int      v4i;

__device__ __forceinline__ int wmma_k(int j, int g) {
  // 16-bit A/B fragment K index for element j of v16h in lane-group g (0/1).
  return (j & 7) + 8 * g + 16 * (j >> 3);
}

__device__ __forceinline__ float lrelu_f(float v) { return v >= 0.f ? v : 0.01f * v; }

// ---------------------------------------------------------------------------
// Tensor Data Mover: load a 2D tile (rows x cols f32, row stride in elements)
// from global memory into LDS at byte offset lds_byte_off.  Descriptor per
// CDNA5 ISA ch.8: group0 = {flags(count=1), lds_addr, global_addr, type=2},
// group1 = {data_size=4B, tensor dims, tile dims, dim0 stride}.  2D tensor ->
// descriptor groups 2/3 (and the trailing group) zero.  Issued per-wave;
// TENSORcnt tracks completion (s_wait_tensorcnt).
// ---------------------------------------------------------------------------
__device__ __forceinline__ void tdm_load_2d(const float* gsrc, unsigned lds_byte_off,
                                            unsigned rows, unsigned cols,
                                            unsigned row_stride_elems) {
  unsigned long long ga = (unsigned long long)(uintptr_t)gsrc;
  v4u g0;
  g0.x = 1u;                                               // count=1 (valid user D#)
  g0.y = lds_byte_off;                                     // lds_addr (bytes)
  g0.z = (unsigned)(ga & 0xFFFFFFFFu);                     // global_addr[31:0]
  g0.w = (unsigned)((ga >> 32) & 0x1FFFFFFu) | (2u << 30); // global_addr[56:32] | type=2
  v8i g1;
  g1[0] = (int)(2u << 16);                                 // wg_mask=0, data_size=2 (4B)
  g1[1] = (int)((cols & 0xFFFFu) << 16);                   // tensor_dim0[15:0]
  g1[2] = (int)(((cols >> 16) & 0xFFFFu) | ((rows & 0xFFFFu) << 16)); // dim0[31:16]|dim1[15:0]
  g1[3] = (int)(((rows >> 16) & 0xFFFFu) | ((cols & 0xFFFFu) << 16)); // dim1[31:16]|tile_dim0
  g1[4] = (int)(rows & 0xFFFFu);                           // tile_dim1 | tile_dim2=0
  g1[5] = (int)row_stride_elems;                           // tensor_dim0_stride[31:0]
  g1[6] = 0;                                               // stride[47:32] | dim1_stride lo
  g1[7] = 0;                                               // dim1_stride hi (unused, 2D)
  v4i zz4 = {0, 0, 0, 0};
  v8i zz8 = {0, 0, 0, 0, 0, 0, 0, 0};
  __builtin_amdgcn_tensor_load_to_lds(g0, g1, zz4, zz4, zz8, 0);
}

// ---------------------------------------------------------------------------
// Per-instance (n,c) mean/var -> affine scale/bias for instance norm fusion.
// ---------------------------------------------------------------------------
__global__ void stats_kernel(const float* __restrict__ x, int C, int HW,
                             const float* __restrict__ g, const float* __restrict__ b,
                             float* __restrict__ scale, float* __restrict__ bias) {
  int inst = blockIdx.x;          // n*C + c
  int c    = inst % C;
  const float* p = x + (size_t)inst * HW;
  __shared__ float ssum[256], ssq[256];
  float s = 0.f, q = 0.f;
  for (int i = threadIdx.x; i < HW; i += blockDim.x) { float v = p[i]; s += v; q += v * v; }
  ssum[threadIdx.x] = s; ssq[threadIdx.x] = q;
  __syncthreads();
  for (int off = 128; off > 0; off >>= 1) {
    if ((int)threadIdx.x < off) { ssum[threadIdx.x] += ssum[threadIdx.x + off]; ssq[threadIdx.x] += ssq[threadIdx.x + off]; }
    __syncthreads();
  }
  if (threadIdx.x == 0) {
    float mu  = ssum[0] / (float)HW;
    float var = ssq[0] / (float)HW - mu * mu;
    float sc  = g[c] * rsqrtf(var + 1e-5f);
    scale[inst] = sc;
    bias[inst]  = b[c] - mu * sc;
  }
}

// ---------------------------------------------------------------------------
// Pack conv weights (OIHW, f32) into f16 laid out in exact WMMA B-fragment
// order: element ((cot*nsteps + ks)*32 + lane)*16 + j  holds
//   W[co0+ (lane&15)][k]  with  k = ks*32 + wmma_k(j, lane>>4),  zero padded.
// The conv inner loop then loads a full B fragment as one v16h vector.
// ---------------------------------------------------------------------------
__global__ void pack_w_kernel(const float* __restrict__ w, _Float16* __restrict__ wpk,
                              int Co, int Ktot, int nsteps) {
  int idx = blockIdx.x * blockDim.x + threadIdx.x;
  int total = ((Co + 15) >> 4) * nsteps * 512;
  if (idx >= total) return;
  int j    = idx & 15;
  int lane = (idx >> 4) & 31;
  int rest = idx >> 9;
  int ks   = rest % nsteps;
  int cot  = rest / nsteps;
  int g = lane >> 4, m = lane & 15;
  int k  = ks * 32 + wmma_k(j, g);
  int co = (cot << 4) + m;
  float v = (co < Co && k < Ktot) ? w[(size_t)co * Ktot + k] : 0.f;
  wpk[idx] = (_Float16)v;
}

// ---------------------------------------------------------------------------
// Generic implicit-GEMM conv on WMMA, specialized on <KHW,KW> so the k->
// (ci,ky,kx) decomposition uses constant divisions.  4 waves per block, each
// wave computes a 16(pixels along W) x 16(output channels) tile; all waves
// share the same packed-weight stream (L0 broadcast).  K swept in 32-steps.
// Optional: per-instance affine on input (fused inorm), leaky-ReLU on input,
// accumulate into out (skip/residual), per-channel output bias.
// ---------------------------------------------------------------------------
template <int KHW, int KW>
__global__ void conv_wmma_kernel(const float* __restrict__ in, const _Float16* __restrict__ wpk,
                                 float* __restrict__ out,
                                 int N, int Ci, int Hi, int Wi, int Co, int Ho, int Wo,
                                 int pad, int stride,
                                 const float* __restrict__ scale, const float* __restrict__ bias,
                                 int act_in, int accum, const float* __restrict__ out_bias) {
  const int lane = threadIdx.x & 31;
  const int wave = threadIdx.x >> 5;
  const int cotiles = (Co + 15) >> 4;
  const int ow0 = ((blockIdx.x << 2) + wave) << 4;
  if (ow0 >= Wo) return;                       // wave-uniform
  const int oh  = blockIdx.y;
  const int cot = blockIdx.z % cotiles;
  const int n   = blockIdx.z / cotiles;
  const int co0 = cot << 4;
  const int g = lane >> 4, m = lane & 15;
  const int Ktot = Ci * KHW;
  const int nsteps = (Ktot + 31) >> 5;

  const v16h* wp = (const v16h*)wpk + (size_t)cot * nsteps * 32 + lane;

  v8f acc = {0.f, 0.f, 0.f, 0.f, 0.f, 0.f, 0.f, 0.f};
  for (int ks = 0; ks < nsteps; ++ks) {
    v16h bfrag = wp[(size_t)ks * 32];          // one 32B vector load
    if (ks + 1 < nsteps)
      __builtin_prefetch((const void*)(wp + (size_t)(ks + 1) * 32), 0, 0);
    v16h afrag;
    const int k0 = ks << 5;
#pragma unroll
    for (int j = 0; j < 16; ++j) {
      int k = k0 + wmma_k(j, g);
      float av = 0.f;
      if (k < Ktot) {
        int ci = k / KHW;                      // constant divisor
        int r  = k - ci * KHW;
        int ky = r / KW;                       // constant divisor
        int kx = r - ky * KW;
        int ih = oh * stride + ky - pad;
        int iw = (ow0 + m) * stride + kx - pad;
        if (ih >= 0 && ih < Hi && iw >= 0 && iw < Wi) {
          av = in[(((size_t)n * Ci + ci) * Hi + ih) * Wi + iw];
          if (scale) { float s = scale[n * Ci + ci]; av = av * s + bias[n * Ci + ci]; }
          if (act_in) av = lrelu_f(av);
        }
      }
      afrag[j] = (_Float16)av;
    }
    acc = __builtin_amdgcn_wmma_f32_16x16x32_f16(false, afrag, false, bfrag, (short)0, acc, false, false);
  }
#pragma unroll
  for (int r = 0; r < 8; ++r) {
    int mm = r + 8 * g;                        // pixel within the 16-wide tile
    int co = co0 + m;                          // output channel
    if (co < Co) {
      size_t idx = (((size_t)n * Co + co) * Ho + oh) * Wo + ow0 + mm;
      float v = acc[r];
      if (out_bias) v += out_bias[co];
      if (accum) v += out[idx];
      out[idx] = v;
    }
  }
}

// ---------------------------------------------------------------------------
// mc_pw with o=h=32: per (n,c) 32x32x32 GEMM  out[o,w] = sum_h W[c,o,h] x[h,w]
// One wave = one 16x16 tile, single K=32 WMMA.
// ---------------------------------------------------------------------------
__global__ void mcpw_wmma_kernel(const float* __restrict__ in, const float* __restrict__ wm,
                                 float* __restrict__ out, int C) {
  const int lane = threadIdx.x;
  const int w0 = (blockIdx.x & 1) << 4;
  const int o0 = ((blockIdx.x >> 1) & 1) << 4;
  const int inst = blockIdx.y;          // n*C + c
  const int c = inst % C;
  const int g = lane >> 4, m = lane & 15;
  v16h afrag, bfrag;
#pragma unroll
  for (int j = 0; j < 16; ++j) {
    int k = wmma_k(j, g);
    afrag[j] = (_Float16)wm[((size_t)c * 32 + o0 + m) * 32 + k];
    bfrag[j] = (_Float16)in[((size_t)inst * 32 + k) * 32 + w0 + m];
  }
  v8f acc = {0.f, 0.f, 0.f, 0.f, 0.f, 0.f, 0.f, 0.f};
  acc = __builtin_amdgcn_wmma_f32_16x16x32_f16(false, afrag, false, bfrag, (short)0, acc, false, false);
#pragma unroll
  for (int r = 0; r < 8; ++r)
    out[((size_t)inst * 32 + o0 + r + 8 * g) * 32 + w0 + m] = acc[r];
}

// ---------------------------------------------------------------------------
// c2dw channel mix: X[n,o,pix] += sum_c W[o,c] * z2[n,c,pix]   (o=64,c=128)
// ---------------------------------------------------------------------------
__global__ void chanmix_wmma_kernel(const float* __restrict__ z2, const float* __restrict__ wt,
                                    float* __restrict__ X) {
  const int lane = threadIdx.x;
  const int g = lane >> 4, m = lane & 15;
  const int pix0 = blockIdx.x << 4;
  const int o0   = blockIdx.y << 4;
  const int n    = blockIdx.z;
  v8f acc = {0.f, 0.f, 0.f, 0.f, 0.f, 0.f, 0.f, 0.f};
  for (int k0 = 0; k0 < 128; k0 += 32) {
    v16h afrag, bfrag;
#pragma unroll
    for (int j = 0; j < 16; ++j) {
      int k = k0 + wmma_k(j, g);
      afrag[j] = (_Float16)wt[(size_t)(o0 + m) * 128 + k];
      bfrag[j] = (_Float16)z2[((size_t)n * 128 + k) * 1024 + pix0 + m];
    }
    acc = __builtin_amdgcn_wmma_f32_16x16x32_f16(false, afrag, false, bfrag, (short)0, acc, false, false);
  }
#pragma unroll
  for (int r = 0; r < 8; ++r) {
    size_t idx = ((size_t)n * 64 + o0 + r + 8 * g) * 1024 + pix0 + m;
    X[idx] += acc[r];
  }
}

// ---------------------------------------------------------------------------
// Generic (small) mc_pw on VALU: out[n,c,o,w] = sum_h W[c,o,h]*act(aff(in))
// blockDim = O*32, grid = N*C.
// ---------------------------------------------------------------------------
__global__ void mcpw_valu_kernel(const float* __restrict__ in, const float* __restrict__ wm,
                                 float* __restrict__ out, int C, int O, int Hd,
                                 const float* __restrict__ scale, const float* __restrict__ bias,
                                 int act_in) {
  int inst = blockIdx.x;          // n*C + c
  int c = inst % C;
  int o = threadIdx.x >> 5, wcol = threadIdx.x & 31;
  float s = scale ? scale[inst] : 1.f;
  float bb = bias ? bias[inst] : 0.f;
  float acc = 0.f;
  for (int h = 0; h < Hd; ++h) {
    float v = in[((size_t)inst * Hd + h) * 32 + wcol] * s + bb;
    if (act_in) v = lrelu_f(v);
    acc += wm[((size_t)c * O + o) * Hd + h] * v;
  }
  out[((size_t)inst * O + o) * 32 + wcol] = acc;
}

// ---------------------------------------------------------------------------
// RoPE (first 8 dims of head-dim 256), in-place on q or k, layout (l,c,32,32).
// ---------------------------------------------------------------------------
__global__ void rope_kernel(float* __restrict__ t) {
  int id = blockIdx.x * blockDim.x + threadIdx.x;      // c*4*16*4 = 16384
  if (id >= 64 * 4 * 16 * 4) return;
  int i    = id & 3;
  int l    = (id >> 2) & 15;
  int head = (id >> 6) & 3;
  int c    = id >> 8;
  float inv_freq = __powf(10000.0f, -(float)(2 * i) / 8.0f);
  float ang = (float)l * inv_freq;
  float cs = __cosf(ang), sn = __sinf(ang);
  size_t base = ((size_t)(l * 64 + c) * 32 + 8 * head) * 32;  // d index = x
  float x1 = t[base + 2 * i], x2 = t[base + 2 * i + 1];
  t[base + 2 * i]     = x1 * cs - x2 * sn;
  t[base + 2 * i + 1] = x2 * cs + x1 * sn;
}

// ---------------------------------------------------------------------------
// qk[c,h,l,m] = (q.k)/sqrt(32) (+prev).  One block per (c,head); 256 threads.
// q(l,d) for fixed (c,head) lives at  l*65536 + c*1024 + head*256 + d :
// a 16 x 256 tile with row stride 65536 -> staged to LDS by the TDM.
// Dynamic LDS: qs @0 (16KB), ks @16384 (16KB).
// ---------------------------------------------------------------------------
__global__ void qk_kernel(const float* __restrict__ q, const float* __restrict__ k,
                          float* __restrict__ qk, int add_prev) {
  int chh = blockIdx.x;             // c*4 + head
  int c = chh >> 2, head = chh & 3;
  extern __shared__ float smem[];
  float* qs = smem;                 // 4096 floats
  float* ks = smem + 4096;          // 4096 floats
  if (threadIdx.x < 32) {           // wave-uniform: wave 0 drives the TDM
    const float* qbase = q + (size_t)c * 1024 + head * 256;
    const float* kbase = k + (size_t)c * 1024 + head * 256;
    tdm_load_2d(qbase, 0u,     16u, 256u, 65536u);
    tdm_load_2d(kbase, 16384u, 16u, 256u, 65536u);
    __builtin_amdgcn_s_wait_tensorcnt(0);
  }
  __syncthreads();
  int l = threadIdx.x >> 4, mcol = threadIdx.x & 15;
  float s = 0.f;
  for (int d = 0; d < 256; ++d) s += qs[l * 256 + d] * ks[mcol * 256 + d];
  s *= 0.17677669529663687f;        // 1/sqrt(32)
  int idx = chh * 256 + l * 16 + mcol;
  if (add_prev) s += qk[idx];
  qk[idx] = s;
}

// ---------------------------------------------------------------------------
// softmax over m then A = P @ V.  One block per (c,head); 256 threads.
// V tile staged to LDS by the TDM (same 16x256 stride-65536 shape).
// Dynamic LDS: vs @0 (16KB), p @16384 (1KB).
// ---------------------------------------------------------------------------
__global__ void attn_av_kernel(const float* __restrict__ qk, const float* __restrict__ v,
                               float* __restrict__ a) {
  int chh = blockIdx.x;
  int c = chh >> 2, head = chh & 3;
  extern __shared__ float smem[];
  float* vs = smem;                 // 4096 floats
  float* p  = smem + 4096;          // 256 floats
  if (threadIdx.x < 32) {
    tdm_load_2d(v + (size_t)c * 1024 + head * 256, 0u, 16u, 256u, 65536u);
    __builtin_amdgcn_s_wait_tensorcnt(0);
  }
  p[threadIdx.x] = qk[chh * 256 + threadIdx.x];
  __syncthreads();
  if (threadIdx.x < 16) {
    int l = threadIdx.x;
    float mx = -1e30f;
    for (int m2 = 0; m2 < 16; ++m2) mx = fmaxf(mx, p[l * 16 + m2]);
    float sum = 0.f;
    for (int m2 = 0; m2 < 16; ++m2) { float e = __expf(p[l * 16 + m2] - mx); p[l * 16 + m2] = e; sum += e; }
    float inv = 1.f / sum;
    for (int m2 = 0; m2 < 16; ++m2) p[l * 16 + m2] *= inv;
  }
  __syncthreads();
  int l = threadIdx.x >> 4, dbase = (threadIdx.x & 15) << 4;
  for (int dd = 0; dd < 16; ++dd) {
    int d = dbase + dd;
    float s = 0.f;
    for (int m2 = 0; m2 < 16; ++m2) s += p[l * 16 + m2] * vs[m2 * 256 + d];
    a[((size_t)(l * 64 + c) * 32 + 8 * head + (d >> 5)) * 32 + (d & 31)] = s;
  }
}

// ---------------------------------------------------------------------------
// Layout shuffles.
// ---------------------------------------------------------------------------
__global__ void to_chunks_kernel(const float* __restrict__ e4, float* __restrict__ ck) {
  int idx = blockIdx.x * blockDim.x + threadIdx.x;   // 64*16384
  if (idx >= 64 * 16 * 32 * 32) return;
  int x = idx & 31, y = (idx >> 5) & 31, c = (idx >> 10) & 63, l = idx >> 16;
  ck[idx] = e4[((size_t)c * 128 + (l >> 2) * 32 + y) * 128 + (l & 3) * 32 + x];
}

__global__ void from_chunks_kernel(const float* __restrict__ ck, float* __restrict__ e4) {
  int idx = blockIdx.x * blockDim.x + threadIdx.x;   // 64*128*128
  if (idx >= 64 * 128 * 128) return;
  int w = idx & 127, h = (idx >> 7) & 127, c = idx >> 14;
  int l = (h >> 5) * 4 + (w >> 5);
  e4[idx] = ck[((size_t)(l * 64 + c) * 32 + (h & 31)) * 32 + (w & 31)];
}

__global__ void concat128_kernel(const float* __restrict__ X, const float* __restrict__ A,
                                 float* __restrict__ z) {
  int idx = blockIdx.x * blockDim.x + threadIdx.x;   // 16*128*1024
  if (idx >= 16 * 128 * 1024) return;
  int pix = idx & 1023, cc = (idx >> 10) & 127, n = idx >> 17;
  z[idx] = (cc < 64) ? X[((size_t)n * 64 + cc) * 1024 + pix]
                     : A[((size_t)n * 64 + cc - 64) * 1024 + pix];
}

__global__ void resize_kernel(const float* __restrict__ in, float* __restrict__ out,
                              int C, int IH, int IW, int OH, int OW) {
  int idx = blockIdx.x * blockDim.x + threadIdx.x;
  if (idx >= C * OH * OW) return;
  int x = idx % OW, y = (idx / OW) % OH, c = idx / (OW * OH);
  float sy = (OH > 1) ? (float)y * (float)(IH - 1) / (float)(OH - 1) : 0.f;
  float sx = (OW > 1) ? (float)x * (float)(IW - 1) / (float)(OW - 1) : 0.f;
  int y0 = (int)floorf(sy); int y1 = min(y0 + 1, IH - 1); float wy = sy - (float)y0;
  int x0 = (int)floorf(sx); int x1 = min(x0 + 1, IW - 1); float wx = sx - (float)x0;
  const float* pc = in + (size_t)c * IH * IW;
  float top = pc[y0 * IW + x0] * (1.f - wx) + pc[y0 * IW + x1] * wx;
  float bot = pc[y1 * IW + x0] * (1.f - wx) + pc[y1 * IW + x1] * wx;
  out[idx] = top * (1.f - wy) + bot * wy;
}

// ---------------------------------------------------------------------------
// Host side orchestration.
// ---------------------------------------------------------------------------
extern "C" void kernel_launch(void* const* d_in, const int* in_sizes, int n_in,
                              void* d_out, int out_size, void* d_ws, size_t ws_size,
                              hipStream_t stream) {
  (void)in_sizes; (void)n_in; (void)out_size; (void)ws_size;
  auto F = [&](int i) { return (const float*)d_in[i]; };

  // ---- param index map (jax pytree: dict keys sorted) ----
  // 0:x  dec1:1..5  dec2:6..10  dec3:11..15  enc1:16..20  enc2:21..25
  // enc3:26..30  enc4:31..35  layers: 36+14*i {c1w,c2dw,c2w,kc,kw,n1b,n1g,
  // n2b,n2g,ow,qc,qw,vc,vw}  148:out_b  149:out_w
  const float* x = F(0);

  // ---- workspace map (floats) ----
  float* W = (float*)d_ws;
  size_t off = 0;
  auto alloc = [&](size_t n) { float* p = W + off; off += (n + 15) & ~(size_t)15; return p; };
  float* e1     = alloc((size_t)8 << 20);
  float* e2     = alloc((size_t)4 << 20);
  float* e3     = alloc((size_t)2 << 20);
  float* e4     = alloc((size_t)1 << 20);
  float* chunks = alloc((size_t)1 << 20);
  float* zbuf   = alloc((size_t)24 << 20);
  float* tbuf   = alloc((size_t)24 << 20);
  float* d3     = alloc((size_t)2 << 20);
  float* d2     = alloc((size_t)4 << 20);
  float* d1     = alloc((size_t)8 << 20);
  float* qtmp   = alloc((size_t)1 << 20);
  float* qb     = alloc((size_t)1 << 20);
  float* kb     = alloc((size_t)1 << 20);
  float* vb     = alloc((size_t)1 << 20);
  float* ab     = alloc((size_t)1 << 20);
  float* ob     = alloc((size_t)1 << 20);
  float* z128   = alloc((size_t)2 << 20);
  float* z1s    = alloc(131072);
  float* z2s    = alloc((size_t)2 << 20);
  float* qkbuf  = alloc(65536);
  float* sc     = alloc(4096);
  float* bs     = alloc(4096);
  _Float16* wpk = (_Float16*)alloc(131072);   // 256K halves, packed conv weights

  auto stats = [&](const float* xin, int N, int C, int HW, const float* g, const float* b) {
    stats_kernel<<<dim3(N * C), dim3(256), 0, stream>>>(xin, C, HW, g, b, sc, bs);
  };
  auto conv = [&](const float* in, const float* wgt, float* out,
                  int N, int Ci, int Hi, int Wi, int Co, int Ho, int Wo,
                  int KH, int KW, int pad, int stride,
                  bool affine, int act_in, int accum, const float* obias) {
    int Ktot = Ci * KH * KW;
    int nsteps = (Ktot + 31) >> 5;
    int cot = (Co + 15) >> 4;
    int total = cot * nsteps * 512;
    pack_w_kernel<<<dim3((total + 255) / 256), dim3(256), 0, stream>>>(wgt, wpk, Co, Ktot, nsteps);
    dim3 grid((Wo + 63) / 64, Ho, N * cot);
    const float* scp = affine ? sc : nullptr;
    const float* bsp = affine ? bs : nullptr;
    if (KW == 3)
      conv_wmma_kernel<9, 3><<<grid, dim3(128), 0, stream>>>(
          in, wpk, out, N, Ci, Hi, Wi, Co, Ho, Wo, pad, stride, scp, bsp, act_in, accum, obias);
    else
      conv_wmma_kernel<1, 1><<<grid, dim3(128), 0, stream>>>(
          in, wpk, out, N, Ci, Hi, Wi, Co, Ho, Wo, pad, stride, scp, bsp, act_in, accum, obias);
  };

  // ================= encoders =================
  auto encoder = [&](const float* xin, int Ci, int Hi, int Co, int s, int pi, float* outb) {
    int Ho = Hi / s;
    stats(xin, 1, Ci, Hi * Hi, F(pi + 1), F(pi + 0));                     // ng, nb
    conv(xin, F(pi + 2), tbuf, 1, Ci, Hi, Hi, Ci, Hi, Hi, 3, 3, 1, 1, true, 0, 0, nullptr);   // w1 on inorm(x)
    conv(xin, F(pi + 4), outb, 1, Ci, Hi, Hi, Co, Ho, Ho, 1, 1, 0, s, false, 0, 0, nullptr);  // skip wi
    conv(tbuf, F(pi + 3), outb, 1, Ci, Hi, Hi, Co, Ho, Ho, 3, 3, 1, s, false, 1, 1, nullptr); // w2(lrelu(t)) +=
  };
  encoder(x,  2, 1024,  8, 1, 16, e1);
  encoder(e1, 8, 1024, 16, 2, 21, e2);
  encoder(e2, 16, 512, 32, 2, 26, e3);
  encoder(e3, 32, 256, 64, 2, 31, e4);

  // ================= transformer =================
  to_chunks_kernel<<<dim3((64 * 16384 + 255) / 256), dim3(256), 0, stream>>>(e4, chunks);

  for (int li = 0; li < 8; ++li) {
    int B = 36 + 14 * li;
    const float *c1w = F(B + 0), *c2dw = F(B + 1), *c2w = F(B + 2);
    const float *kc = F(B + 3), *kw = F(B + 4);
    const float *n1b = F(B + 5), *n1g = F(B + 6), *n2b = F(B + 7), *n2g = F(B + 8);
    const float *ow = F(B + 9), *qc = F(B + 10), *qw = F(B + 11), *vc = F(B + 12), *vw = F(B + 13);

    stats(chunks, 16, 64, 1024, n1g, n1b);
    // q/k/v = mc_pw(conv3x3(inorm(x)), *)
    conv(chunks, qc, qtmp, 16, 64, 32, 32, 64, 32, 32, 3, 3, 1, 1, true, 0, 0, nullptr);
    mcpw_wmma_kernel<<<dim3(4, 16 * 64), dim3(32), 0, stream>>>(qtmp, qw, qb, 64);
    conv(chunks, kc, qtmp, 16, 64, 32, 32, 64, 32, 32, 3, 3, 1, 1, true, 0, 0, nullptr);
    mcpw_wmma_kernel<<<dim3(4, 16 * 64), dim3(32), 0, stream>>>(qtmp, kw, kb, 64);
    conv(chunks, vc, qtmp, 16, 64, 32, 32, 64, 32, 32, 3, 3, 1, 1, true, 0, 0, nullptr);
    mcpw_wmma_kernel<<<dim3(4, 16 * 64), dim3(32), 0, stream>>>(qtmp, vw, vb, 64);

    rope_kernel<<<dim3(64), dim3(256), 0, stream>>>(qb);
    rope_kernel<<<dim3(64), dim3(256), 0, stream>>>(kb);
    qk_kernel<<<dim3(256), dim3(256), 32768, stream>>>(qb, kb, qkbuf, li > 0 ? 1 : 0);
    attn_av_kernel<<<dim3(256), dim3(256), (4096 + 256) * 4, stream>>>(qkbuf, vb, ab);
    mcpw_wmma_kernel<<<dim3(4, 16 * 64), dim3(32), 0, stream>>>(ab, ow, ob, 64);

    concat128_kernel<<<dim3((16 * 128 * 1024 + 255) / 256), dim3(256), 0, stream>>>(chunks, ob, z128);
    stats(z128, 16, 128, 1024, n2g, n2b);
    mcpw_valu_kernel<<<dim3(16 * 128), dim3(2 * 32), 0, stream>>>(z128, c1w, z1s, 128, 2, 32, sc, bs, 0);
    mcpw_valu_kernel<<<dim3(16 * 128), dim3(32 * 32), 0, stream>>>(z1s, c2w, z2s, 128, 32, 2, nullptr, nullptr, 1);
    chanmix_wmma_kernel<<<dim3(64, 4, 16), dim3(32), 0, stream>>>(z2s, c2dw, chunks); // residual +=
  }

  from_chunks_kernel<<<dim3((64 * 128 * 128 + 255) / 256), dim3(256), 0, stream>>>(chunks, e4);

  // ================= decoders =================
  auto decoder = [&](const float* xin, int Cx, int Hx, const float* skip, int Cs, int Hs,
                     int Co, int pi, float* outb) {
    int C = Cs + Cx, H = Hs;
    (void)hipMemcpyAsync(zbuf, skip, (size_t)Cs * H * H * sizeof(float), hipMemcpyDeviceToDevice, stream);
    resize_kernel<<<dim3((Cx * H * H + 255) / 256), dim3(256), 0, stream>>>(
        xin, zbuf + (size_t)Cs * H * H, Cx, Hx, Hx, H, H);
    stats(zbuf, 1, C, H * H, F(pi + 1), F(pi + 0));
    conv(zbuf, F(pi + 4), outb, 1, C, H, H, Co, H, H, 1, 1, 0, 1, false, 0, 0, nullptr);  // wi(z)
    conv(zbuf, F(pi + 2), tbuf, 1, C, H, H, C, H, H, 3, 3, 1, 1, true, 0, 0, nullptr);    // w1(inorm(z))
    conv(tbuf, F(pi + 3), outb, 1, C, H, H, Co, H, H, 3, 3, 1, 1, false, 1, 1, nullptr);  // w2(lrelu) +=
  };
  decoder(e4, 64, 128, e3, 32, 256, 32, 11, d3);
  decoder(d3, 32, 256, e2, 16, 512, 16,  6, d2);
  decoder(d2, 16, 512, e1,  8, 1024, 8,  1, d1);

  // ================= output head =================
  conv(d1, F(149), (float*)d_out, 1, 8, 1024, 1024, 2, 1024, 1024, 1, 1, 0, 1,
       false, 0, 0, F(148));
}
